// MinamoModel2_82987358093801
// MI455X (gfx1250) — compile-verified, matrix operand order checked by reading, not run on hardware
//
#include <hip/hip_runtime.h>
#include <cmath>

// ---------------------------------------------------------------------------
// MI455X (gfx1250). wave32, WMMA bf16 16x16x32, f32 accumulate.
// Fast path: 64x128 block tile, 8 waves as 2(M)x4(N), wave tile 32x32
// -> 4 v_wmma per K-step, zero bounds checks (shapes guaranteed aligned).
// Generic guarded kernel kept only for the 3 odd shapes (K=16, K=1, N=32).
// LDS row stride 40 bf16 (80B) keeps all fragment chunks 16B-aligned:
// staging = ds_store_b128, fragment reads = ds_load_b128.
// ---------------------------------------------------------------------------

typedef __bf16 bf16_t;
typedef __attribute__((ext_vector_type(16))) bf16_t v16bf;
typedef __attribute__((ext_vector_type(8)))  bf16_t v8bf;
typedef __attribute__((ext_vector_type(8)))  float  v8f;

constexpr int NBATCH = 128;
constexpr int HW     = 169;          // 13*13
constexpr int NP     = NBATCH * HW;  // 21632 rows (multiple of 64)

#define BK 32
#define LSTR 40   // LDS row stride in bf16 elements (80 bytes, 16B aligned)

enum { ACT_NONE = 0, ACT_GELU = 1, ACT_RELU = 2 };

__device__ __forceinline__ float gelu_f(float x) {
  return 0.5f * x * (1.0f + erff(x * 0.70710678118654752f));
}

__device__ __forceinline__ float act_f(float v, int act) {
  if (act == ACT_GELU) return gelu_f(v);
  if (act == ACT_RELU) return fmaxf(v, 0.0f);
  return v;
}

__device__ __forceinline__ v8bf pack8(float a0, float a1, float a2, float a3,
                                      float a4, float a5, float a6, float a7) {
  v8bf p;
  p[0] = (bf16_t)a0; p[1] = (bf16_t)a1; p[2] = (bf16_t)a2; p[3] = (bf16_t)a3;
  p[4] = (bf16_t)a4; p[5] = (bf16_t)a5; p[6] = (bf16_t)a6; p[7] = (bf16_t)a7;
  return p;
}

__device__ __forceinline__ v8bf pack8v(float4 f0, float4 f1) {
  return pack8(f0.x, f0.y, f0.z, f0.w, f1.x, f1.y, f1.z, f1.w);
}

__device__ __forceinline__ v16bf frag16(const bf16_t* base, int koff) {
  // lane fragment = two contiguous 16B chunks: k in [koff,koff+8) / [koff+16,koff+24)
  v8bf lo = *(const v8bf*)(base + koff);
  v8bf hi = *(const v8bf*)(base + koff + 16);
  return __builtin_shufflevector(lo, hi, 0, 1, 2, 3, 4, 5, 6, 7,
                                 8, 9, 10, 11, 12, 13, 14, 15);
}

#define WMMA_BF16(a, b, c) \
  __builtin_amdgcn_wmma_f32_16x16x32_bf16(false, (a), false, (b), (short)0, (c), false, false)

// ---------------------------------------------------------------------------
// FAST GEMM: requires M%64==0, N%128==0, K%32==0 (checked by host dispatch).
// C[m, coff+n] = act(A[M,K] @ W[N,K]^T + bias [+ residual]); no guards at all.
// ---------------------------------------------------------------------------
__global__ __launch_bounds__(256) void gemm_wmma_fast(
    const float* __restrict__ A, int lda,
    const float* __restrict__ W,            // [N, K] row-major
    const float* __restrict__ bias,
    const float* __restrict__ residual, int ldr,
    float* __restrict__ C, int ldc, int coff,
    int K, int act)
{
  __shared__ bf16_t As[64 * LSTR];
  __shared__ bf16_t Bs[128 * LSTR];

  const int tid  = threadIdx.x;
  const int lane = tid & 31;
  const int l15  = lane & 15;
  const int wv   = tid >> 5;
  const int wm   = wv & 1;     // 2 waves along M (32 rows each)
  const int wn   = wv >> 1;    // 4 waves along N (32 cols each)
  const int m0   = blockIdx.y * 64;
  const int n0   = blockIdx.x * 128;

  const int lrowA = tid >> 2, lkcA = (tid & 3) * 8;   // 8 A elems/thread
  const int lrowB = tid >> 1, lkcB = (tid & 1) * 16;  // 16 B elems/thread

  v8f acc00 = {}, acc01 = {}, acc10 = {}, acc11 = {};

  const float* aptr = A + (long)(m0 + lrowA) * lda + lkcA;
  const float* wptr = W + (long)(n0 + lrowB) * K + lkcB;

  for (int k0 = 0; k0 < K; k0 += BK) {
    {
      const float* ap = aptr + k0;
      *(v8bf*)&As[lrowA * LSTR + lkcA] =
          pack8v(*(const float4*)ap, *(const float4*)(ap + 4));
    }
    {
      const float* wp = wptr + k0;
      *(v8bf*)&Bs[lrowB * LSTR + lkcB] =
          pack8v(*(const float4*)wp, *(const float4*)(wp + 4));
      *(v8bf*)&Bs[lrowB * LSTR + lkcB + 8] =
          pack8v(*(const float4*)(wp + 8), *(const float4*)(wp + 12));
    }
    __syncthreads();

    const int koff = (lane < 16) ? 0 : 8;
    const bf16_t* ar = &As[(wm * 32 + l15) * LSTR];
    const bf16_t* br = &Bs[(wn * 32 + l15) * LSTR];
    v16bf a0 = frag16(ar, koff);
    v16bf a1 = frag16(ar + 16 * LSTR, koff);
    v16bf b0 = frag16(br, koff);
    v16bf b1 = frag16(br + 16 * LSTR, koff);
    acc00 = WMMA_BF16(a0, b0, acc00);
    acc01 = WMMA_BF16(a0, b1, acc01);
    acc10 = WMMA_BF16(a1, b0, acc10);
    acc11 = WMMA_BF16(a1, b1, acc11);
    __syncthreads();
  }

  const int nn0 = n0 + wn * 32 + l15;
  const int nn1 = nn0 + 16;
  const float bi0 = bias ? bias[nn0] : 0.0f;
  const float bi1 = bias ? bias[nn1] : 0.0f;
#pragma unroll
  for (int r = 0; r < 8; r++) {
    int mm0 = m0 + wm * 32 + r + ((lane < 16) ? 0 : 8);
    int mm1 = mm0 + 16;
    float v00 = acc00[r] + bi0, v01 = acc01[r] + bi1;
    float v10 = acc10[r] + bi0, v11 = acc11[r] + bi1;
    if (residual) {
      v00 += residual[(long)mm0 * ldr + nn0];
      v01 += residual[(long)mm0 * ldr + nn1];
      v10 += residual[(long)mm1 * ldr + nn0];
      v11 += residual[(long)mm1 * ldr + nn1];
    }
    C[(long)mm0 * ldc + coff + nn0] = act_f(v00, act);
    C[(long)mm0 * ldc + coff + nn1] = act_f(v01, act);
    C[(long)mm1 * ldc + coff + nn0] = act_f(v10, act);
    C[(long)mm1 * ldc + coff + nn1] = act_f(v11, act);
  }
}

// ---------------------------------------------------------------------------
// Generic guarded GEMM (only for K=16, K=1, N=32 odd shapes). 64x64 tile.
// ---------------------------------------------------------------------------
__global__ __launch_bounds__(256) void gemm_wmma_gen(
    const float* __restrict__ A, int lda,
    const float* __restrict__ W,
    const float* __restrict__ bias,
    float* __restrict__ C, int ldc, int coff,
    int M, int N, int K, int act)
{
  __shared__ bf16_t As[64 * LSTR];
  __shared__ bf16_t Bs[64 * LSTR];

  const int tid  = threadIdx.x;
  const int lane = tid & 31;
  const int l15  = lane & 15;
  const int wv   = tid >> 5;
  const int wm   = wv & 3;
  const int wn   = wv >> 2;
  const int m0   = blockIdx.y * 64;
  const int n0   = blockIdx.x * 64;

  const int lrow = tid >> 2;
  const int lkc  = (tid & 3) * 8;

  v8f acc0 = {}, acc1 = {};

  for (int k0 = 0; k0 < K; k0 += BK) {
    {
      int gm = m0 + lrow, gk0 = k0 + lkc;
      v8bf pk;
      if (gm < M && gk0 + 8 <= K) {
        const float* ap = A + (long)gm * lda + gk0;
        pk = pack8v(*(const float4*)ap, *(const float4*)(ap + 4));
      } else {
        int gmc = gm < M ? gm : (M - 1);
#pragma unroll
        for (int j = 0; j < 8; j++) {
          int gk = gk0 + j;
          int gkc = gk < K ? gk : (K - 1);
          float v = A[(long)gmc * lda + gkc];
          pk[j] = (bf16_t)((gm < M && gk < K) ? v : 0.0f);
        }
      }
      *(v8bf*)&As[lrow * LSTR + lkc] = pk;
    }
    {
      int gn = n0 + lrow, gk0 = k0 + lkc;
      v8bf pk;
      if (gn < N && gk0 + 8 <= K) {
        const float* wp = W + (long)gn * K + gk0;
        pk = pack8v(*(const float4*)wp, *(const float4*)(wp + 4));
      } else {
        int gnc = gn < N ? gn : (N - 1);
#pragma unroll
        for (int j = 0; j < 8; j++) {
          int gk = gk0 + j;
          int gkc = gk < K ? gk : (K - 1);
          float v = W[(long)gnc * K + gkc];
          pk[j] = (bf16_t)((gn < N && gk < K) ? v : 0.0f);
        }
      }
      *(v8bf*)&Bs[lrow * LSTR + lkc] = pk;
    }
    __syncthreads();

    const int koff = (lane < 16) ? 0 : 8;
    const bf16_t* ar = &As[(wm * 16 + l15) * LSTR];
    const bf16_t* br = &Bs[(wn * 32 + l15) * LSTR];
    v16bf a  = frag16(ar, koff);
    v16bf b0 = frag16(br, koff);
    v16bf b1 = frag16(br + 16 * LSTR, koff);
    acc0 = WMMA_BF16(a, b0, acc0);
    acc1 = WMMA_BF16(a, b1, acc1);
    __syncthreads();
  }

  const int nn0 = n0 + wn * 32 + l15;
  const int nn1 = nn0 + 16;
#pragma unroll
  for (int r = 0; r < 8; r++) {
    int mm = m0 + wm * 16 + r + ((lane < 16) ? 0 : 8);
    if (mm < M) {
      if (nn0 < N)
        C[(long)mm * ldc + coff + nn0] = act_f(acc0[r] + (bias ? bias[nn0] : 0.0f), act);
      if (nn1 < N)
        C[(long)mm * ldc + coff + nn1] = act_f(acc1[r] + (bias ? bias[nn1] : 0.0f), act);
    }
  }
}

// ---------------------------------------------------------------------------
// Implicit-GEMM 3x3 conv, replicate padding, tap-major reduction order
// (k = tap*Cin + ci). M = NP (always aligned), only N may be odd (Cout=32).
// 64x128 tile, wave tile 32x32, 4 WMMAs / K-step. A side branch-free.
// ---------------------------------------------------------------------------
__global__ __launch_bounds__(256) void conv_wmma_bf16(
    const float* __restrict__ X, int Cin,   // NHWC rows [NP, Cin]
    const float* __restrict__ W,            // [Cout, Cin*9] OIHW-flat
    const float* __restrict__ bias,
    float* __restrict__ C, int ldc, int coff,
    int N, int act)
{
  const int K = Cin * 9;
  __shared__ bf16_t As[64 * LSTR];
  __shared__ bf16_t Bs[128 * LSTR];

  const int tid  = threadIdx.x;
  const int lane = tid & 31;
  const int l15  = lane & 15;
  const int wv   = tid >> 5;
  const int wm   = wv & 1;
  const int wn   = wv >> 1;
  const int m0   = blockIdx.y * 64;
  const int n0   = blockIdx.x * 128;

  const int lrowA = tid >> 2, lkcA = (tid & 3) * 8;
  const int lrowB = tid >> 1, lkcB = (tid & 1) * 16;

  // fixed pixel decode for this thread's A row
  const int gm   = m0 + lrowA;            // always < NP
  const int bimg = gm / HW;
  const int pix  = gm - bimg * HW;
  const int pr   = pix / 13;
  const int pcc  = pix - pr * 13;
  const int gn   = n0 + lrowB;

  v8f acc00 = {}, acc01 = {}, acc10 = {}, acc11 = {};

  for (int k0 = 0; k0 < K; k0 += BK) {
    { // A: on-the-fly im2col; chunk never crosses a tap (Cin%8==0)
      int gk0 = k0 + lkcA;
      int tap = gk0 / Cin;
      int ci0 = gk0 - tap * Cin;
      int dr = tap / 3 - 1, dc = tap % 3 - 1;
      int rr = pr + dr, cc = pcc + dc;
      rr = rr < 0 ? 0 : (rr > 12 ? 12 : rr);
      cc = cc < 0 ? 0 : (cc > 12 ? 12 : cc);
      const float* src = X + (long)(bimg * HW + rr * 13 + cc) * Cin + ci0;
      *(v8bf*)&As[lrowA * LSTR + lkcA] =
          pack8v(*(const float4*)src, *(const float4*)(src + 4));
    }
    { // B: remap tap-major k -> ci*9+tap (stride-9 gather; chunk Cin%16==0)
      int gk0 = k0 + lkcB;
      int tap = gk0 / Cin;
      int ci0 = gk0 - tap * Cin;
      v8bf p0, p1;
      if (gn < N) {
        const float* wr = W + (long)gn * K + (long)ci0 * 9 + tap;
#pragma unroll
        for (int j = 0; j < 8; j++) p0[j] = (bf16_t)wr[j * 9];
#pragma unroll
        for (int j = 0; j < 8; j++) p1[j] = (bf16_t)wr[(j + 8) * 9];
      } else {
#pragma unroll
        for (int j = 0; j < 8; j++) { p0[j] = (bf16_t)0.0f; p1[j] = (bf16_t)0.0f; }
      }
      *(v8bf*)&Bs[lrowB * LSTR + lkcB]     = p0;
      *(v8bf*)&Bs[lrowB * LSTR + lkcB + 8] = p1;
    }
    __syncthreads();

    const int koff = (lane < 16) ? 0 : 8;
    const bf16_t* ar = &As[(wm * 32 + l15) * LSTR];
    const bf16_t* br = &Bs[(wn * 32 + l15) * LSTR];
    v16bf a0 = frag16(ar, koff);
    v16bf a1 = frag16(ar + 16 * LSTR, koff);
    v16bf b0 = frag16(br, koff);
    v16bf b1 = frag16(br + 16 * LSTR, koff);
    acc00 = WMMA_BF16(a0, b0, acc00);
    acc01 = WMMA_BF16(a0, b1, acc01);
    acc10 = WMMA_BF16(a1, b0, acc10);
    acc11 = WMMA_BF16(a1, b1, acc11);
    __syncthreads();
  }

  const int nn0 = n0 + wn * 32 + l15;
  const int nn1 = nn0 + 16;
  const float bi0 = nn0 < N ? bias[nn0] : 0.0f;
  const float bi1 = nn1 < N ? bias[nn1] : 0.0f;
#pragma unroll
  for (int r = 0; r < 8; r++) {
    int mm0 = m0 + wm * 32 + r + ((lane < 16) ? 0 : 8);
    int mm1 = mm0 + 16;
    if (nn0 < N) {
      C[(long)mm0 * ldc + coff + nn0] = act_f(acc00[r] + bi0, act);
      C[(long)mm1 * ldc + coff + nn0] = act_f(acc10[r] + bi0, act);
    }
    if (nn1 < N) {
      C[(long)mm0 * ldc + coff + nn1] = act_f(acc01[r] + bi1, act);
      C[(long)mm1 * ldc + coff + nn1] = act_f(acc11[r] + bi1, act);
    }
  }
}

// ---------------------------------------------------------------------------
// Small / irregular kernels (<<1% of FLOPs)
// ---------------------------------------------------------------------------
__global__ void nchw_to_rows_k(const float* __restrict__ x, float* __restrict__ y,
                               int B, int C) {
  int i = blockIdx.x * blockDim.x + threadIdx.x;
  if (i >= B * C * HW) return;
  int b = i / (C * HW), rem = i % (C * HW);
  int c = rem / HW, p = rem % HW;
  y[(long)(b * HW + p) * C + c] = x[i];
}

__global__ void grid_attn_k(const float* __restrict__ q, const float* __restrict__ kmat,
                            const float* __restrict__ v, const float* __restrict__ skip,
                            float* __restrict__ out, int ldo, int ooff,
                            int heads, int dh) {
  int gid = blockIdx.x * blockDim.x + threadIdx.x;
  if (gid >= NP * heads) return;
  int n = gid / heads, h = gid % heads;
  int b = n / HW, p = n % HW, r = p / 13, c = p % 13;
  int D = heads * dh;
  const float* qn = q + (long)n * D + h * dh;
  float scale = rsqrtf((float)dh);

  float lg[9]; int nbr[9]; int cnt = 0; float mx = -1e30f;
  for (int dr = -1; dr <= 1; dr++)
    for (int dc = -1; dc <= 1; dc++) {
      int rr = r + dr, cc = c + dc;
      if (rr < 0 || rr > 12 || cc < 0 || cc > 12) continue;
      int src = b * HW + rr * 13 + cc;
      const float* kn = kmat + (long)src * D + h * dh;
      float d = 0.0f;
      for (int t = 0; t < dh; t++) d += qn[t] * kn[t];
      d *= scale;
      lg[cnt] = d; nbr[cnt] = src;
      if (d > mx) mx = d;
      cnt++;
    }
  float s = 0.0f;
  for (int e = 0; e < cnt; e++) { lg[e] = __expf(lg[e] - mx); s += lg[e]; }
  float inv = 1.0f / s;
  for (int t = 0; t < dh; t++) {
    float a = 0.0f;
    for (int e = 0; e < cnt; e++) a += lg[e] * v[(long)nbr[e] * D + h * dh + t];
    float o = a * inv + skip[(long)n * D + h * dh + t];
    out[(long)n * ldo + ooff + h * dh + t] = gelu_f(o);
  }
}

__global__ void enc_attn_k(const float* __restrict__ qkv, float* __restrict__ out, int B) {
  int gid = blockIdx.x * blockDim.x + threadIdx.x;
  if (gid >= B * 24) return;
  int b = gid / 24, rem = gid % 24, h = rem / 3, sq = rem % 3;
  const int dh = 32;
  const float* q = qkv + (long)(b * 3 + sq) * 768 + h * dh;
  float lg[3], mx = -1e30f;
  for (int sk = 0; sk < 3; sk++) {
    const float* kk = qkv + (long)(b * 3 + sk) * 768 + 256 + h * dh;
    float d = 0.0f;
    for (int t = 0; t < dh; t++) d += q[t] * kk[t];
    lg[sk] = d * 0.17677669529663689f;
    if (lg[sk] > mx) mx = lg[sk];
  }
  float s = 0.0f;
  for (int sk = 0; sk < 3; sk++) { lg[sk] = __expf(lg[sk] - mx); s += lg[sk]; }
  float inv = 1.0f / s;
  for (int t = 0; t < dh; t++) {
    float a = 0.0f;
    for (int sk = 0; sk < 3; sk++)
      a += lg[sk] * qkv[(long)(b * 3 + sk) * 768 + 512 + h * dh + t];
    out[(long)(b * 3 + sq) * 256 + h * dh + t] = a * inv;
  }
}

__global__ void layernorm_k(const float* __restrict__ x, const float* __restrict__ g,
                            const float* __restrict__ b, float* __restrict__ y,
                            int rows, int D) {
  int r = blockIdx.x * blockDim.x + threadIdx.x;
  if (r >= rows) return;
  const float* xr = x + (long)r * D;
  float m = 0.0f;
  for (int i = 0; i < D; i++) m += xr[i];
  m /= D;
  float var = 0.0f;
  for (int i = 0; i < D; i++) { float d = xr[i] - m; var += d * d; }
  var /= D;
  float inv = rsqrtf(var + 1e-5f);
  for (int i = 0; i < D; i++)
    y[(long)r * D + i] = (xr[i] - m) * inv * g[i] + b[i];
}

__global__ void mean3_k(const float* __restrict__ feat, float* __restrict__ mf, int B) {
  int i = blockIdx.x * blockDim.x + threadIdx.x;
  if (i >= B * 256) return;
  int b = i / 256, c = i % 256;
  float s = feat[(long)(b * 3 + 0) * 256 + c] + feat[(long)(b * 3 + 1) * 256 + c] +
            feat[(long)(b * 3 + 2) * 256 + c];
  mf[i] = s * (1.0f / 3.0f);
}

__global__ void fill_stage_k(float* __restrict__ st, const int* __restrict__ stage, int B) {
  int b = blockIdx.x * blockDim.x + threadIdx.x;
  if (b < B) st[b] = (float)stage[0];
}

__global__ void inject_k(float* __restrict__ h, const float* __restrict__ g,
                         const float* __restrict__ bb, int C) {
  int i = blockIdx.x * blockDim.x + threadIdx.x;
  if (i >= NP * C) return;
  int n = i / C, c = i % C, b = n / HW;
  h[i] = h[i] * g[(long)b * C + c] + bb[(long)b * C + c];
}

__global__ void maxpool_k(const float* __restrict__ x, float* __restrict__ pooled, int C) {
  int i = blockIdx.x * blockDim.x + threadIdx.x;
  if (i >= NBATCH * C) return;
  int b = i / C, c = i % C;
  float m = -1e30f;
  for (int p = 0; p < HW; p++) {
    float v = x[(long)(b * HW + p) * C + c];
    m = v > m ? v : m;
  }
  pooled[i] = m;
}

__global__ void head_final_k(const float* __restrict__ pooled, const float* __restrict__ pc,
                             const float* __restrict__ Wfc, const float* __restrict__ bfc,
                             float* __restrict__ out, int C) {
  int b = blockIdx.x * blockDim.x + threadIdx.x;
  if (b >= NBATCH) return;
  float s = bfc[0];
  for (int c = 0; c < C; c++) {
    float p = pooled[(long)b * C + c];
    s += p * Wfc[c] + p * pc[(long)b * C + c];
  }
  out[b] = s;
}

// ---------------------------------------------------------------------------
// Host side
// ---------------------------------------------------------------------------
static inline int nb256(long n) { return (int)((n + 255) / 256); }

static void gemm(hipStream_t st, const float* A, int lda, const float* W,
                 const float* bias, const float* res, int ldr,
                 float* C, int ldc, int coff, int M, int N, int K, int act) {
  if ((M & 63) == 0 && (N & 127) == 0 && (K & 31) == 0) {
    dim3 g(N / 128, M / 64);
    gemm_wmma_fast<<<g, 256, 0, st>>>(A, lda, W, bias, res, ldr, C, ldc, coff, K, act);
  } else {
    dim3 g((N + 63) / 64, (M + 63) / 64);
    gemm_wmma_gen<<<g, 256, 0, st>>>(A, lda, W, bias, C, ldc, coff, M, N, K, act);
  }
}

static void convg(hipStream_t st, const float* X, int Cin, const float* W,
                  const float* bias, float* C, int ldc, int coff, int Cout, int act) {
  dim3 g((Cout + 127) / 128, NP / 64);
  conv_wmma_bf16<<<g, 256, 0, st>>>(X, Cin, W, bias, C, ldc, coff, Cout, act);
}

struct TCp { const float *Wq, *Wk, *Wv, *Ws, *bq, *bk, *bv, *bs; };
struct CFp {
  const float *c1W, *c1b, *c2W, *c2b;
  const float *f1W, *f1b, *f2W, *f2b;
  TCp tc1, tc2;
};

#define PF(i) ((const float*)d_in[(i)])

static CFp getCF(void* const* d_in, int base) {
  CFp p;
  p.c1W = PF(base + 0);  p.c1b = PF(base + 1);
  p.c2W = PF(base + 2);  p.c2b = PF(base + 3);
  p.f1W = PF(base + 4);  p.f1b = PF(base + 5);
  p.f2W = PF(base + 6);  p.f2b = PF(base + 7);
  p.tc1 = { PF(base + 9),  PF(base + 8),  PF(base + 11), PF(base + 10),
            PF(base + 13), PF(base + 12), PF(base + 15), PF(base + 14) };
  p.tc2 = { PF(base + 17), PF(base + 16), PF(base + 19), PF(base + 18),
            PF(base + 21), PF(base + 20), PF(base + 23), PF(base + 22) };
  return p;
}

// workspace offsets (floats)
static const size_t o_x    = 0;
static const size_t o_h    = o_x    + (size_t)NP * 32;
static const size_t o_h2   = o_h    + (size_t)NP * 256;
static const size_t o_t1   = o_h2   + (size_t)NP * 256;
static const size_t o_cat  = o_t1   + (size_t)NP * 512;
static const size_t o_q    = o_cat  + (size_t)NP * 512;
static const size_t o_k    = o_q    + (size_t)NP * 512;
static const size_t o_v    = o_k    + (size_t)NP * 512;
static const size_t o_sk   = o_v    + (size_t)NP * 512;
static const size_t o_a1   = o_sk   + (size_t)NP * 512;
static const size_t o_sm   = o_a1   + (size_t)NP * 512;
static const size_t o_st   = o_sm;
static const size_t o_tmp  = o_st   + 128;
static const size_t o_feat = o_tmp  + 384 * 256;
static const size_t o_qkv  = o_feat + 384 * 256;
static const size_t o_attn = o_qkv  + 384 * 768;
static const size_t o_ff   = o_attn + 384 * 256;
static const size_t o_mf   = o_ff   + 384 * 1024;
static const size_t o_cond = o_mf   + 128 * 256;
static const size_t o_g    = o_cond + 128 * 256;
static const size_t o_b    = o_g    + 128 * 256;
static const size_t o_pc   = o_b    + 128 * 256;
static const size_t o_pool = o_pc   + 128 * 256;

static void conv_fusion(hipStream_t s, float* w, const float* X,
                        int inc, int hid, int out, const CFp& p, float* outbuf) {
  float* t1  = w + o_t1;
  float* cat = w + o_cat;
  float* q   = w + o_q;
  float* k   = w + o_k;
  float* v   = w + o_v;
  float* sk  = w + o_sk;
  float* a1  = w + o_a1;

  convg(s, X, inc, p.c1W, p.c1b, t1, hid, 0, hid, ACT_GELU);
  convg(s, t1, hid, p.c2W, p.c2b, cat, 2 * inc, 0, inc, ACT_GELU);

  gemm(s, X, inc, p.tc1.Wq, p.tc1.bq, nullptr, 0, q,  hid, 0, NP, hid, inc, ACT_NONE);
  gemm(s, X, inc, p.tc1.Wk, p.tc1.bk, nullptr, 0, k,  hid, 0, NP, hid, inc, ACT_NONE);
  gemm(s, X, inc, p.tc1.Wv, p.tc1.bv, nullptr, 0, v,  hid, 0, NP, hid, inc, ACT_NONE);
  gemm(s, X, inc, p.tc1.Ws, p.tc1.bs, nullptr, 0, sk, hid, 0, NP, hid, inc, ACT_NONE);
  grid_attn_k<<<nb256((long)NP * 8), 256, 0, s>>>(q, k, v, sk, a1, hid, 0, 8, hid / 8);

  gemm(s, a1, hid, p.tc2.Wq, p.tc2.bq, nullptr, 0, q,  inc, 0, NP, inc, hid, ACT_NONE);
  gemm(s, a1, hid, p.tc2.Wk, p.tc2.bk, nullptr, 0, k,  inc, 0, NP, inc, hid, ACT_NONE);
  gemm(s, a1, hid, p.tc2.Wv, p.tc2.bv, nullptr, 0, v,  inc, 0, NP, inc, hid, ACT_NONE);
  gemm(s, a1, hid, p.tc2.Ws, p.tc2.bs, nullptr, 0, sk, inc, 0, NP, inc, hid, ACT_NONE);
  grid_attn_k<<<nb256((long)NP), 256, 0, s>>>(q, k, v, sk, cat, 2 * inc, inc, 1, inc);

  convg(s, cat, 2 * inc, p.f1W, p.f1b, t1, hid, 0, hid, ACT_GELU);
  convg(s, t1, hid, p.f2W, p.f2b, outbuf, out, 0, out, ACT_GELU);
}

extern "C" void kernel_launch(void* const* d_in, const int* in_sizes, int n_in,
                              void* d_out, int out_size, void* d_ws, size_t ws_size,
                              hipStream_t stream) {
  (void)in_sizes; (void)n_in; (void)out_size; (void)ws_size;
  const float* x        = PF(0);
  const float* tag_cond = PF(1);
  const float* val_cond = PF(2);
  const int*   stage    = (const int*)d_in[3];
  float* w = (float*)d_ws;
  hipStream_t s = stream;

  const int PB = 4;
  const float *cfL1W = PF(PB + 0), *cfL1b = PF(PB + 1), *cfL2W = PF(PB + 2), *cfL2b = PF(PB + 3);
  const int SB = PB + 52, TB = PB + 56, VB = PB + 60;
  const int C1 = PB + 64, C2 = PB + 88, C3 = PB + 112, HB = PB + 136;
  const float *fcW = PF(HB + 24), *fcb = PF(HB + 25);
  const float *projW = PF(HB + 26), *projb = PF(HB + 27);
  const float *injbW = PF(PB + 164), *injbb = PF(PB + 165);
  const float *injgW = PF(PB + 166), *injgb = PF(PB + 167);

  // ================= condition encoder =================
  fill_stage_k<<<1, 128, 0, s>>>(w + o_st, stage, NBATCH);

  gemm(s, tag_cond, 64, PF(TB + 0), PF(TB + 1), nullptr, 0, w + o_tmp, 256, 0, NBATCH, 256, 64, ACT_GELU);
  gemm(s, w + o_tmp, 256, PF(TB + 2), PF(TB + 3), nullptr, 0, w + o_feat, 768, 0, NBATCH, 256, 256, ACT_GELU);
  gemm(s, val_cond, 16, PF(VB + 0), PF(VB + 1), nullptr, 0, w + o_tmp, 256, 0, NBATCH, 256, 16, ACT_GELU);
  gemm(s, w + o_tmp, 256, PF(VB + 2), PF(VB + 3), nullptr, 0, w + o_feat, 768, 256, NBATCH, 256, 256, ACT_GELU);
  gemm(s, w + o_st, 1, PF(SB + 0), PF(SB + 1), nullptr, 0, w + o_tmp, 256, 0, NBATCH, 256, 1, ACT_GELU);
  gemm(s, w + o_tmp, 256, PF(SB + 2), PF(SB + 3), nullptr, 0, w + o_feat, 768, 512, NBATCH, 256, 256, ACT_GELU);

  for (int li = 0; li < 4; li++) {
    const int LB = PB + 4 + 12 * li;
    const float *Wo = PF(LB + 0), *Wqkv = PF(LB + 1), *bo = PF(LB + 2), *bqkv = PF(LB + 3);
    const float *f1W = PF(LB + 4), *f1b = PF(LB + 5), *f2W = PF(LB + 6), *f2b = PF(LB + 7);
    const float *ln1b = PF(LB + 8), *ln1g = PF(LB + 9), *ln2b = PF(LB + 10), *ln2g = PF(LB + 11);

    gemm(s, w + o_feat, 256, Wqkv, bqkv, nullptr, 0, w + o_qkv, 768, 0, 384, 768, 256, ACT_NONE);
    enc_attn_k<<<nb256(NBATCH * 24), 256, 0, s>>>(w + o_qkv, w + o_attn, NBATCH);
    gemm(s, w + o_attn, 256, Wo, bo, w + o_feat, 256, w + o_tmp, 256, 0, 384, 256, 256, ACT_NONE);
    layernorm_k<<<nb256(384), 256, 0, s>>>(w + o_tmp, ln1g, ln1b, w + o_feat, 384, 256);
    gemm(s, w + o_feat, 256, f1W, f1b, nullptr, 0, w + o_ff, 1024, 0, 384, 1024, 256, ACT_RELU);
    gemm(s, w + o_ff, 1024, f2W, f2b, w + o_feat, 256, w + o_tmp, 256, 0, 384, 256, 1024, ACT_NONE);
    layernorm_k<<<nb256(384), 256, 0, s>>>(w + o_tmp, ln2g, ln2b, w + o_feat, 384, 256);
  }

  mean3_k<<<nb256(NBATCH * 256), 256, 0, s>>>(w + o_feat, w + o_mf, NBATCH);
  gemm(s, w + o_mf, 256, cfL1W, cfL1b, nullptr, 0, w + o_tmp, 256, 0, NBATCH, 256, 256, ACT_GELU);
  gemm(s, w + o_tmp, 256, cfL2W, cfL2b, nullptr, 0, w + o_cond, 256, 0, NBATCH, 256, 256, ACT_NONE);

  gemm(s, w + o_cond, 256, injgW, injgb, nullptr, 0, w + o_g, 256, 0, NBATCH, 256, 256, ACT_NONE);
  gemm(s, w + o_cond, 256, injbW, injbb, nullptr, 0, w + o_b, 256, 0, NBATCH, 256, 256, ACT_NONE);

  // ================= conv backbone =================
  nchw_to_rows_k<<<nb256((long)NBATCH * 32 * HW), 256, 0, s>>>(x, w + o_x, NBATCH, 32);

  CFp cf1 = getCF(d_in, C1), cf2 = getCF(d_in, C2), cf3 = getCF(d_in, C3), cfh = getCF(d_in, HB);
  conv_fusion(s, w, w + o_x,  32,  256, 256, cf1, w + o_h);
  conv_fusion(s, w, w + o_h,  256, 512, 256, cf2, w + o_h2);
  conv_fusion(s, w, w + o_h2, 256, 512, 256, cf3, w + o_h);

  inject_k<<<nb256((long)NP * 256), 256, 0, s>>>(w + o_h, w + o_g, w + o_b, 256);

  // ================= head (stage 0) =================
  conv_fusion(s, w, w + o_h, 256, 256, 256, cfh, w + o_h2);
  maxpool_k<<<nb256(NBATCH * 256), 256, 0, s>>>(w + o_h2, w + o_pool, 256);
  gemm(s, w + o_cond, 256, projW, projb, nullptr, 0, w + o_pc, 256, 0, NBATCH, 256, 256, ACT_NONE);
  head_final_k<<<(NBATCH + 63) / 64, 64, 0, s>>>(w + o_pool, w + o_pc, fcW, fcb,
                                                 (float*)d_out, 256);
}
